// TransformerBlock_68968584839184
// MI455X (gfx1250) — matrix operand order repaired and checked by reading, not compiled, not run on hardware
//
#include <hip/hip_runtime.h>
#include <hip/hip_bf16.h>
#include <stdint.h>

typedef __attribute__((ext_vector_type(16))) _Float16 v16h;
typedef __attribute__((ext_vector_type(8)))  float    v8f;
typedef __attribute__((ext_vector_type(4))) unsigned int u32x4;
typedef __attribute__((ext_vector_type(8))) int          i32x8;
typedef __attribute__((ext_vector_type(4))) int          i32x4;

#define Bz   2
#define NT   2048
#define Cd   1024
#define Hh   16
#define Dh   64
#define FFd  4096
#define Mrow 4096   // B*NT

// ---------------------------------------------------------------------------
// Tensor Data Mover helpers (descriptor per CDNA5 ISA 08_async_tensor 8.3/8.4)
// ---------------------------------------------------------------------------
__device__ __forceinline__ void tdm_load(u32x4 g0, i32x8 g1) {
    i32x4 z4 = {0, 0, 0, 0};
#if __clang_major__ >= 23
    i32x8 z8 = {0, 0, 0, 0, 0, 0, 0, 0};
    __builtin_amdgcn_tensor_load_to_lds(g0, g1, z4, z4, z8, 0);
#else
    __builtin_amdgcn_tensor_load_to_lds(g0, g1, z4, z4, 0);
#endif
}

// group0: [1:0]=count=1 | [63:32]=lds_addr | [120:64]=global_addr | [127:126]=type=2
__device__ __forceinline__ u32x4 tdm_g0(uint32_t lds_off, uint64_t gaddr) {
    u32x4 g0;
    g0.x = 1u;
    g0.y = lds_off;
    g0.z = (uint32_t)gaddr;
    g0.w = (uint32_t)(gaddr >> 32) | (2u << 30);
    return g0;
}

// group1 for a 2D f16 tile: tile_d0 x tile_d1 elements, row stride `stride0`
// elements, LDS padding: +1 DWORD after every 16 DWORDs (pitch 17 per 64B).
__device__ __forceinline__ i32x8 tdm_g1_2d(uint32_t tens_d0, uint32_t tens_d1,
                                           uint32_t tile_d0, uint32_t tile_d1,
                                           uint32_t stride0) {
    i32x8 g1;
    g1[0] = (int)((1u << 16)        // data_size = 1 -> 2 bytes
                | (1u << 20)        // pad_enable
                | (3u << 22)        // pad_interval code 3 = 16 DWORDs
                | (0u << 25));      // pad_amount  code 0 = 1 DWORD
    g1[1] = (int)((tens_d0 & 0xffffu) << 16);
    g1[2] = (int)((tens_d0 >> 16) | ((tens_d1 & 0xffffu) << 16));
    g1[3] = (int)((tens_d1 >> 16) | (tile_d0 << 16));
    g1[4] = (int)(tile_d1 & 0xffffu);     // tile_dim1 ; tile_dim2 = 0
    g1[5] = (int)stride0;                 // tensor_dim0_stride[31:0]
    g1[6] = 0;                            // stride[47:32]=0 ; dim1_stride lo=0
    g1[7] = 0;
    return g1;
}

// ---------------------------------------------------------------------------
// Weight transpose + f32->f16 convert:  wT[n][k] = (f16) w[k][n]
// ---------------------------------------------------------------------------
__global__ __launch_bounds__(256)
void transpose_cvt_kernel(const float* __restrict__ w, _Float16* __restrict__ wT,
                          int K, int N)
{
    __shared__ float tile[32][33];
    const int n0 = blockIdx.x * 32;
    const int k0 = blockIdx.y * 32;
    const int tx = threadIdx.x & 31;
    const int ty = threadIdx.x >> 5;       // 0..7
#pragma unroll
    for (int i = 0; i < 4; i++) {
        int k = ty + i * 8;
        tile[k][tx] = w[(size_t)(k0 + k) * N + n0 + tx];
    }
    __syncthreads();
#pragma unroll
    for (int i = 0; i < 4; i++) {
        int n = ty + i * 8;
        wT[(size_t)(n0 + n) * K + k0 + tx] = (_Float16)tile[tx][n];
    }
}

// ---------------------------------------------------------------------------
// LayerNorm (f32 in) -> f16 out, one row (C=1024) per 256-thread block
// ---------------------------------------------------------------------------
__global__ __launch_bounds__(256)
void layernorm_kernel(const float* __restrict__ x, const float* __restrict__ g,
                      const float* __restrict__ b, _Float16* __restrict__ out)
{
    __shared__ float red[2][8];
    const int row = blockIdx.x;
    const float* xr = x + (size_t)row * Cd;
    float v[4], s = 0.f, s2 = 0.f;
#pragma unroll
    for (int i = 0; i < 4; i++) {
        v[i] = xr[threadIdx.x + i * 256];
        s += v[i]; s2 += v[i] * v[i];
    }
#pragma unroll
    for (int off = 1; off < 32; off <<= 1) {
        s  += __shfl_xor(s,  off, 32);
        s2 += __shfl_xor(s2, off, 32);
    }
    const int wave = threadIdx.x >> 5, lane = threadIdx.x & 31;
    if (lane == 0) { red[0][wave] = s; red[1][wave] = s2; }
    __syncthreads();
    s = 0.f; s2 = 0.f;
#pragma unroll
    for (int i = 0; i < 8; i++) { s += red[0][i]; s2 += red[1][i]; }
    const float mu   = s * (1.0f / Cd);
    const float var  = s2 * (1.0f / Cd) - mu * mu;
    const float rstd = rsqrtf(var + 1e-5f);
#pragma unroll
    for (int i = 0; i < 4; i++) {
        int c = threadIdx.x + i * 256;
        out[(size_t)row * Cd + c] = (_Float16)((v[i] - mu) * rstd * g[c] + b[c]);
    }
}

// ---------------------------------------------------------------------------
// WMMA GEMM with TDM double-buffered tile staging.
// C[M,N] = epilogue( A[M,K] @ Bt[N,K]^T )
// MODE 0: plain f16 store; MODE 1: f32 store + bias + residual;
// MODE 2: f16 store of gelu(acc + bias)
// Block = 256 thr (8 waves), tile 128x128, wave tile 64x32, K-step 64.
// TDM pads LDS rows to 34 u32 (two conflict-free pitch-17 half-tiles).
// ---------------------------------------------------------------------------
union Frag { v16h h; uint32_t u[8]; };

template<int MODE>
__global__ __launch_bounds__(256)
void gemm_f16(const _Float16* __restrict__ A, const _Float16* __restrict__ Bt,
              _Float16* __restrict__ Ch, float* __restrict__ Cf,
              const float* __restrict__ bias, const float* __restrict__ res,
              int Ndim, int K)
{
    __shared__ uint32_t sA[2][128 * 34];
    __shared__ uint32_t sB[2][128 * 34];
    const int tid  = threadIdx.x;
    const int lane = tid & 31;
    const int wave = tid >> 5;
    const int half = lane >> 4;
    const int l16  = lane & 15;
    const int m0   = blockIdx.y * 128;
    const int n0   = blockIdx.x * 128;
    const int wm   = (wave & 1) * 64;
    const int wn   = (wave >> 1) * 32;

    v8f acc[4][2];
    const v8f vzero = {0.f,0.f,0.f,0.f,0.f,0.f,0.f,0.f};
#pragma unroll
    for (int i = 0; i < 4; i++)
#pragma unroll
        for (int j = 0; j < 2; j++) acc[i][j] = vzero;

    const uint64_t aBase = (uint64_t)(const void*)(A  + (size_t)m0 * K);
    const uint64_t bBase = (uint64_t)(const void*)(Bt + (size_t)n0 * K);
    const uint32_t ldsA0 = (uint32_t)(uintptr_t)&sA[0][0];
    const uint32_t ldsA1 = (uint32_t)(uintptr_t)&sA[1][0];
    const uint32_t ldsB0 = (uint32_t)(uintptr_t)&sB[0][0];
    const uint32_t ldsB1 = (uint32_t)(uintptr_t)&sB[1][0];
    const i32x8 g1 = tdm_g1_2d(/*tens_d0=*/64, /*tens_d1=*/128,
                               /*tile_d0=*/64, /*tile_d1=*/128,
                               /*stride0=*/(uint32_t)K);

    if (wave == 0) {
        tdm_load(tdm_g0(ldsA0, aBase), g1);
        tdm_load(tdm_g0(ldsB0, bBase), g1);
    }

    const int nk = K >> 6;
    for (int it = 0; it < nk; ++it) {
        const int cur = it & 1;
        if (wave == 0) __builtin_amdgcn_s_wait_tensorcnt(0);
        __syncthreads();                       // buf[cur] ready; buf[cur^1] free
        if (wave == 0 && (it + 1) < nk) {
            uint64_t off = (uint64_t)(it + 1) * 128;   // 64 f16 = 128 bytes
            tdm_load(tdm_g0(cur ? ldsA0 : ldsA1, aBase + off), g1);
            tdm_load(tdm_g0(cur ? ldsB0 : ldsB1, bBase + off), g1);
        }
        const uint32_t* __restrict__ tA = sA[cur];
        const uint32_t* __restrict__ tB = sB[cur];
#pragma unroll
        for (int ks = 0; ks < 2; ks++) {       // two 32-wide K slices
            Frag a[4], b[2];
#pragma unroll
            for (int mt = 0; mt < 4; mt++) {
                int row = wm + mt * 16 + l16;
#pragma unroll
                for (int v = 0; v < 8; v++) {
                    int g = v >> 2, j = v & 3;
                    a[mt].u[v] = tA[row * 34 + ks * 17 + g * 8 + half * 4 + j];
                }
            }
#pragma unroll
            for (int nt = 0; nt < 2; nt++) {
                int row = wn + nt * 16 + l16;
#pragma unroll
                for (int v = 0; v < 8; v++)
                    b[nt].u[v] = tB[row * 34 + ks * 17 + (v >> 2) * 8 + half * 4 + (v & 3)];
            }
#pragma unroll
            for (int mt = 0; mt < 4; mt++)
#pragma unroll
                for (int nt = 0; nt < 2; nt++)
                    acc[mt][nt] = __builtin_amdgcn_wmma_f32_16x16x32_f16(
                        false, a[mt].h, false, b[nt].h, (short)0, acc[mt][nt], false, false);
        }
    }

#pragma unroll
    for (int mt = 0; mt < 4; mt++) {
#pragma unroll
        for (int nt = 0; nt < 2; nt++) {
#pragma unroll
            for (int j = 0; j < 8; j++) {
                int row = m0 + wm + mt * 16 + j + half * 8;
                int col = n0 + wn + nt * 16 + l16;
                float val = acc[mt][nt][j];
                if constexpr (MODE == 0) {
                    Ch[(size_t)row * Ndim + col] = (_Float16)val;
                } else if constexpr (MODE == 1) {
                    Cf[(size_t)row * Ndim + col] =
                        val + bias[col] + res[(size_t)row * Ndim + col];
                } else {
                    float xb = val + bias[col];
                    float gl = 0.5f * xb * (1.0f + erff(xb * 0.70710678f));
                    Ch[(size_t)row * Ndim + col] = (_Float16)gl;
                }
            }
        }
    }
}

// ---------------------------------------------------------------------------
// qkv [M, 3C] f16 -> Q[BH,N,D], K[BH,N,D], Vt[BH,D,N]
// ---------------------------------------------------------------------------
__global__ __launch_bounds__(256)
void qkv_split_kernel(const _Float16* __restrict__ qkv, _Float16* __restrict__ Q,
                      _Float16* __restrict__ Kk, _Float16* __restrict__ Vt)
{
    size_t idx = (size_t)blockIdx.x * 256 + threadIdx.x;   // < 4096*3072
    int col = (int)(idx % (3 * Cd));
    int row = (int)(idx / (3 * Cd));
    int b = row >> 11, n = row & 2047;
    int which = col >> 10;
    int h = (col >> 6) & 15;
    int d = col & 63;
    _Float16 val = qkv[idx];
    size_t bh = (size_t)(b * Hh + h);
    if (which == 0)       Q [(bh * NT + n) * Dh + d] = val;
    else if (which == 1)  Kk[(bh * NT + n) * Dh + d] = val;
    else                  Vt[(bh * Dh + d) * NT + n] = val;
}

// ---------------------------------------------------------------------------
// Flash attention. One block = 8 waves, all on the same (b,h); each wave owns
// 16 query rows. K (64x64) and Vt (64x64) tiles are staged into LDS once per
// block via TDM (double-buffered, wave 0 issues DMA), instead of 8x redundant
// per-wave global loads. Online softmax; S=Q@K^T and O+=P@V via WMMA.
// O written as [B,N,H,D] f16.
// ---------------------------------------------------------------------------
__global__ __launch_bounds__(256)
void attention_kernel(const _Float16* __restrict__ Q, const _Float16* __restrict__ Kg,
                      const _Float16* __restrict__ Vt, _Float16* __restrict__ O)
{
    __shared__ uint32_t sK[2][64 * 34];                 // padded 64x64 f16 tiles
    __shared__ uint32_t sV[2][64 * 34];
    __shared__ __align__(16) _Float16 sP[8][16 * 64];   // per-wave P staging
    const int tid  = threadIdx.x;
    const int lane = tid & 31;
    const int wave = tid >> 5;
    const int half = lane >> 4;
    const int l16  = lane & 15;

    const int bh = blockIdx.x >> 4;                     // 16 blocks per (b,h)
    const int q0 = ((blockIdx.x & 15) * 8 + wave) * 16; // this wave's 16 rows

    const uint32_t* Qu = (const uint32_t*)(Q + (size_t)bh * NT * Dh);

    // Q fragments (16 rows x D=64 -> two K-steps), row pitch 32 u32
    Frag qa[2];
#pragma unroll
    for (int ks = 0; ks < 2; ks++)
#pragma unroll
        for (int v = 0; v < 8; v++) {
            int g = v >> 2, j = v & 3;
            qa[ks].u[v] = Qu[(size_t)(q0 + l16) * 32 + ks * 16 + g * 8 + half * 4 + j];
        }

    const v8f vzero = {0.f,0.f,0.f,0.f,0.f,0.f,0.f,0.f};
    float mrun[8], lrun[8];
    v8f oacc[4];
#pragma unroll
    for (int j = 0; j < 8; j++) { mrun[j] = -1e30f; lrun[j] = 0.f; }
#pragma unroll
    for (int t = 0; t < 4; t++) oacc[t] = vzero;

    _Float16* pbuf  = &sP[wave][0];
    uint32_t* pbu32 = (uint32_t*)pbuf;

    // TDM staging of K / Vt tiles
    const uint64_t kBase = (uint64_t)(const void*)(Kg + (size_t)bh * NT * Dh);
    const uint64_t vBase = (uint64_t)(const void*)(Vt + (size_t)bh * Dh * NT);
    const uint32_t ldsK0 = (uint32_t)(uintptr_t)&sK[0][0];
    const uint32_t ldsK1 = (uint32_t)(uintptr_t)&sK[1][0];
    const uint32_t ldsV0 = (uint32_t)(uintptr_t)&sV[0][0];
    const uint32_t ldsV1 = (uint32_t)(uintptr_t)&sV[1][0];
    const i32x8 g1K = tdm_g1_2d(64, 64, 64, 64, /*stride0=*/Dh);  // K rows: pitch 64
    const i32x8 g1V = tdm_g1_2d(64, 64, 64, 64, /*stride0=*/NT);  // Vt rows: pitch 2048

    if (wave == 0) {
        tdm_load(tdm_g0(ldsK0, kBase), g1K);
        tdm_load(tdm_g0(ldsV0, vBase), g1V);
    }

    for (int key0 = 0; key0 < NT; key0 += 64) {
        const int cur = (key0 >> 6) & 1;
        if (wave == 0) __builtin_amdgcn_s_wait_tensorcnt(0);
        __syncthreads();                   // tiles[cur] ready; tiles[cur^1] free
        if (wave == 0 && (key0 + 64) < NT) {
            tdm_load(tdm_g0(cur ? ldsK0 : ldsK1, kBase + (uint64_t)(key0 + 64) * Dh * 2), g1K);
            tdm_load(tdm_g0(cur ? ldsV0 : ldsV1, vBase + (uint64_t)(key0 + 64) * 2),      g1V);
        }
        const uint32_t* __restrict__ tK = sK[cur];
        const uint32_t* __restrict__ tV = sV[cur];

        // ---- S = (Q @ K^T) * scale ----
        v8f s[4];
#pragma unroll
        for (int nt = 0; nt < 4; nt++) {
            s[nt] = vzero;
#pragma unroll
            for (int ks = 0; ks < 2; ks++) {
                Frag bf;
                int rowb = nt * 16 + l16;
#pragma unroll
                for (int v = 0; v < 8; v++)
                    bf.u[v] = tK[rowb * 34 + ks * 17 + (v >> 2) * 8 + half * 4 + (v & 3)];
                s[nt] = __builtin_amdgcn_wmma_f32_16x16x32_f16(
                    false, qa[ks].h, false, bf.h, (short)0, s[nt], false, false);
            }
        }
        // ---- online softmax ----
        float alpha[8];
#pragma unroll
        for (int j = 0; j < 8; j++) {
            float mx = -1e30f;
#pragma unroll
            for (int nt = 0; nt < 4; nt++) {
                s[nt][j] *= 0.125f;              // 1/sqrt(64)
                mx = fmaxf(mx, s[nt][j]);
            }
#pragma unroll
            for (int off = 1; off < 16; off <<= 1)
                mx = fmaxf(mx, __shfl_xor(mx, off, 32));
            float mnew = fmaxf(mrun[j], mx);
            alpha[j] = __expf(mrun[j] - mnew);
            mrun[j] = mnew;
            float rs = 0.f;
#pragma unroll
            for (int nt = 0; nt < 4; nt++) {
                float p = __expf(s[nt][j] - mnew);
                s[nt][j] = p;
                rs += p;
            }
#pragma unroll
            for (int off = 1; off < 16; off <<= 1)
                rs += __shfl_xor(rs, off, 32);
            lrun[j] = lrun[j] * alpha[j] + rs;
#pragma unroll
            for (int t = 0; t < 4; t++) oacc[t][j] *= alpha[j];
        }
        // ---- stage P (16x64) into LDS as f16 ----
#pragma unroll
        for (int nt = 0; nt < 4; nt++)
#pragma unroll
            for (int j = 0; j < 8; j++)
                pbuf[(j + 8 * half) * 64 + nt * 16 + l16] = (_Float16)s[nt][j];
        asm volatile("s_wait_dscnt 0" ::: "memory");
        // ---- O += P @ V ----
        Frag pa[2];
#pragma unroll
        for (int ks = 0; ks < 2; ks++)
#pragma unroll
            for (int v = 0; v < 8; v++) {
                int g = v >> 2, j = v & 3;
                pa[ks].u[v] = pbu32[l16 * 32 + ks * 16 + g * 8 + half * 4 + j];
            }
#pragma unroll
        for (int nt = 0; nt < 4; nt++) {       // d tiles
#pragma unroll
            for (int ks = 0; ks < 2; ks++) {
                Frag bf;
                int rowd = nt * 16 + l16;
#pragma unroll
                for (int v = 0; v < 8; v++)
                    bf.u[v] = tV[rowd * 34 + ks * 17 + (v >> 2) * 8 + half * 4 + (v & 3)];
                oacc[nt] = __builtin_amdgcn_wmma_f32_16x16x32_f16(
                    false, pa[ks].h, false, bf.h, (short)0, oacc[nt], false, false);
            }
        }
        asm volatile("s_wait_dscnt 0" ::: "memory");
    }
    // ---- epilogue: divide by l, store to [B,N,H,D] ----
    const int b = bh >> 4, h = bh & 15;
#pragma unroll
    for (int j = 0; j < 8; j++) {
        float inv = 1.0f / lrun[j];
        int n = q0 + j + 8 * half;
#pragma unroll
        for (int nt = 0; nt < 4; nt++) {
            int d = nt * 16 + l16;
            O[(((size_t)b * NT + n) * Hh + h) * Dh + d] = (_Float16)(oacc[nt][j] * inv);
        }
    }
}

// ---------------------------------------------------------------------------
// Host-side launcher
// ---------------------------------------------------------------------------
extern "C" void kernel_launch(void* const* d_in, const int* in_sizes, int n_in,
                              void* d_out, int out_size, void* d_ws, size_t ws_size,
                              hipStream_t stream)
{
    const float* x      = (const float*)d_in[0];
    const float* ln1_g  = (const float*)d_in[1];
    const float* ln1_b  = (const float*)d_in[2];
    const float* w_qkv  = (const float*)d_in[3];
    const float* w_proj = (const float*)d_in[4];
    const float* b_proj = (const float*)d_in[5];
    const float* ln2_g  = (const float*)d_in[6];
    const float* ln2_b  = (const float*)d_in[7];
    const float* w_fc1  = (const float*)d_in[8];
    const float* b_fc1  = (const float*)d_in[9];
    const float* w_fc2  = (const float*)d_in[10];
    const float* b_fc2  = (const float*)d_in[11];
    float* out = (float*)d_out;

    char* ws = (char*)d_ws;
    const size_t MB = 1ull << 20;
    _Float16* h1     = (_Float16*)(ws + 0);          //  8 MB  [4096,1024]
    _Float16* qkv    = (_Float16*)(ws + 8 * MB);     // 24 MB  [4096,3072]
    _Float16* hf     = (_Float16*)(ws + 0);          // 32 MB  [4096,4096] (aliases h1+qkv)
    _Float16* Qb     = (_Float16*)(ws + 32 * MB);    //  8 MB  [32,2048,64]
    _Float16* Kb     = (_Float16*)(ws + 40 * MB);    //  8 MB
    _Float16* Vtb    = (_Float16*)(ws + 48 * MB);    //  8 MB  [32,64,2048]
    _Float16* ob     = (_Float16*)(ws + 56 * MB);    //  8 MB  [4096,1024]
    float*    x2     = (float*)   (ws + 64 * MB);    // 16 MB  [4096,1024]
    _Float16* h2     = (_Float16*)(ws + 80 * MB);    //  8 MB
    _Float16* wqkvT  = (_Float16*)(ws + 88 * MB);    //  6 MB  [3072,1024]
    _Float16* wprojT = (_Float16*)(ws + 94 * MB);    //  2 MB  [1024,1024]
    _Float16* wfc1T  = (_Float16*)(ws + 96 * MB);    //  8 MB  [4096,1024]
    _Float16* wfc2T  = (_Float16*)(ws + 104 * MB);   //  8 MB  [1024,4096]

    // 1. weight transpose/convert
    transpose_cvt_kernel<<<dim3(3 * Cd / 32, Cd / 32),  256, 0, stream>>>(w_qkv,  wqkvT,  Cd,  3 * Cd);
    transpose_cvt_kernel<<<dim3(Cd / 32,     Cd / 32),  256, 0, stream>>>(w_proj, wprojT, Cd,  Cd);
    transpose_cvt_kernel<<<dim3(FFd / 32,    Cd / 32),  256, 0, stream>>>(w_fc1,  wfc1T,  Cd,  FFd);
    transpose_cvt_kernel<<<dim3(Cd / 32,     FFd / 32), 256, 0, stream>>>(w_fc2,  wfc2T,  FFd, Cd);

    // 2. LN1
    layernorm_kernel<<<Mrow, 256, 0, stream>>>(x, ln1_g, ln1_b, h1);

    // 3. QKV gemm: [4096,1024]@[1024,3072] -> f16
    gemm_f16<0><<<dim3(3 * Cd / 128, Mrow / 128), 256, 0, stream>>>(
        h1, wqkvT, qkv, nullptr, nullptr, nullptr, 3 * Cd, Cd);

    // 4. split into Q, K, Vt
    qkv_split_kernel<<<(Mrow * 3 * Cd) / 256, 256, 0, stream>>>(qkv, Qb, Kb, Vtb);

    // 5. attention -> ob [4096,1024] f16
    attention_kernel<<<512, 256, 0, stream>>>(Qb, Kb, Vtb, ob);

    // 6. proj gemm + bias + residual -> x2 (f32)
    gemm_f16<1><<<dim3(Cd / 128, Mrow / 128), 256, 0, stream>>>(
        ob, wprojT, nullptr, x2, b_proj, x, Cd, Cd);

    // 7. LN2
    layernorm_kernel<<<Mrow, 256, 0, stream>>>(x2, ln2_g, ln2_b, h2);

    // 8. FC1 gemm + bias + gelu -> hf f16
    gemm_f16<2><<<dim3(FFd / 128, Mrow / 128), 256, 0, stream>>>(
        h2, wfc1T, hf, nullptr, b_fc1, nullptr, FFd, Cd);

    // 9. FC2 gemm + bias + residual -> d_out (f32)
    gemm_f16<1><<<dim3(Cd / 128, Mrow / 128), 256, 0, stream>>>(
        hf, wfc2T, nullptr, out, b_fc2, x2, Cd, FFd);

    (void)in_sizes; (void)n_in; (void)out_size; (void)ws_size;
}
